// MoE_66752381715016
// MI455X (gfx1250) — matrix operand order repaired and checked by reading, not compiled, not run on hardware
//
#include <hip/hip_runtime.h>
#include <hip/hip_bf16.h>
#include <math.h>

#define BB   8
#define NN   2048
#define DD   1024
#define EE   8
#define HH   4096
#define CAP  320
#define MROWS (BB * CAP)   // 2560 rows per expert

typedef __attribute__((ext_vector_type(16))) __bf16 v16bf;
typedef __attribute__((ext_vector_type(8)))  float  v8f;

// ---------------------------------------------------------------------------
// 0) init: zero loss accumulators, slot_token = -1
// ---------------------------------------------------------------------------
__global__ __launch_bounds__(256) void init_kernel(float* __restrict__ sumraw,
                                                   int* __restrict__ slot_token) {
    int t = blockIdx.x * blockDim.x + threadIdx.x;
    if (t < BB * EE) sumraw[t] = 0.0f;
    for (int i = t; i < EE * BB * CAP; i += gridDim.x * blockDim.x)
        slot_token[i] = -1;
}

// ---------------------------------------------------------------------------
// 1) gating: one wave32 per token
// ---------------------------------------------------------------------------
__global__ __launch_bounds__(256) void gating_kernel(
    const float* __restrict__ x, const float* __restrict__ wg,
    const float* __restrict__ rp,
    int* __restrict__ e1o, int* __restrict__ e2o,
    float* __restrict__ g1o, float* __restrict__ g2o,
    int* __restrict__ keep2o, int* __restrict__ slot1, int* __restrict__ slot2,
    float* __restrict__ sumraw)
{
    const int wave = threadIdx.x >> 5;
    const int lane = threadIdx.x & 31;
    const int tok  = blockIdx.x * 8 + wave;

    const float* xr = x + (size_t)tok * DD;
    float acc[8] = {0.f, 0.f, 0.f, 0.f, 0.f, 0.f, 0.f, 0.f};

    for (int i = 0; i < DD / 32; ++i) {
        const int d = lane + i * 32;
        const float xv = xr[d];
        const float4* w4 = (const float4*)(wg + (size_t)d * EE);
        const float4 wa = w4[0];
        const float4 wb = w4[1];
        acc[0] += xv * wa.x; acc[1] += xv * wa.y;
        acc[2] += xv * wa.z; acc[3] += xv * wa.w;
        acc[4] += xv * wb.x; acc[5] += xv * wb.y;
        acc[6] += xv * wb.z; acc[7] += xv * wb.w;
    }
#pragma unroll
    for (int e = 0; e < 8; ++e)
#pragma unroll
        for (int o = 16; o >= 1; o >>= 1)
            acc[e] += __shfl_xor(acc[e], o, 32);

    // softmax (redundant on all lanes; lane 0 writes)
    float mx = acc[0];
#pragma unroll
    for (int e = 1; e < 8; ++e) mx = fmaxf(mx, acc[e]);
    float p[8], s = 0.f;
#pragma unroll
    for (int e = 0; e < 8; ++e) { p[e] = expf(acc[e] - mx); s += p[e]; }
    const float inv = 1.0f / s;
#pragma unroll
    for (int e = 0; e < 8; ++e) p[e] *= inv;

    // top-1 (first-occurrence argmax)
    int i1 = 0; float v1 = p[0];
#pragma unroll
    for (int e = 1; e < 8; ++e) { if (p[e] > v1) { v1 = p[e]; i1 = e; } }
    // top-2 among e != i1 (first occurrence)
    int i2 = (i1 == 0) ? 1 : 0; float v2 = p[i2];
#pragma unroll
    for (int e = 0; e < 8; ++e) {
        if (e != i1 && p[e] > v2) { v2 = p[e]; i2 = e; }
    }

    const float denom = v1 + v2 + 1e-9f;
    const float G1 = v1 / denom;
    const float G2 = v2 / denom;
    const int kp = (rp[tok] < (G2 / 0.2f)) ? 1 : 0;  // random-second policy

    if (lane == 0) {
        e1o[tok] = i1; e2o[tok] = i2;
        g1o[tok] = G1; g2o[tok] = G2;
        keep2o[tok] = kp;
        slot1[tok] = -1; slot2[tok] = -1;
        const int b = tok >> 11;  // tok / NN
#pragma unroll
        for (int e = 0; e < 8; ++e)
            atomicAdd(&sumraw[b * EE + e], p[e]);
    }
}

// ---------------------------------------------------------------------------
// 2) scan: per (b,e) sequential cumsum with capacity truncation
// ---------------------------------------------------------------------------
__global__ void scan_kernel(const int* __restrict__ e1, const int* __restrict__ e2,
                            const int* __restrict__ keep2,
                            int* __restrict__ slot1, int* __restrict__ slot2,
                            int* __restrict__ slot_token, int* __restrict__ cnt1tot)
{
    const int t = threadIdx.x;
    if (t >= BB * EE) return;
    const int b = t / EE, e = t % EE;
    const int base  = b * NN;
    const int sbase = (e * BB + b) * CAP;

    int cpre = 0;
    for (int n = 0; n < NN; ++n) {
        if (e1[base + n] == e) {
            if (cpre < CAP) {
                slot1[base + n] = sbase + cpre;
                slot_token[sbase + cpre] = base + n;
            }
            cpre++;
        }
    }
    cnt1tot[b * EE + e] = cpre;                 // pre-truncation count for loss
    const int m1 = (cpre < CAP) ? cpre : CAP;   // truncated mask1 count
    int c2 = 0;
    for (int n = 0; n < NN; ++n) {
        if (e2[base + n] == e && keep2[base + n]) {
            const int pos = m1 + c2;
            if (pos < CAP) {
                slot2[base + n] = sbase + pos;
                slot_token[sbase + pos] = base + n;
            }
            c2++;
        }
    }
}

// ---------------------------------------------------------------------------
// 3) dispatch: gather x rows -> bf16 capacity buffers
// ---------------------------------------------------------------------------
__global__ __launch_bounds__(256) void dispatch_kernel(
    const float* __restrict__ x, const int* __restrict__ slot_token,
    __bf16* __restrict__ ei)
{
    const int slot = blockIdx.x;
    const int tok = slot_token[slot];
    if (tok < 0) return;   // uniform per block
    const float4* src = (const float4*)(x + (size_t)tok * DD);
    __bf16* dst = ei + (size_t)slot * DD;
    const int i = threadIdx.x;     // DD/4 == 256
    const float4 v = src[i];
    dst[i * 4 + 0] = (__bf16)v.x;
    dst[i * 4 + 1] = (__bf16)v.y;
    dst[i * 4 + 2] = (__bf16)v.z;
    dst[i * 4 + 3] = (__bf16)v.w;
}

// ---------------------------------------------------------------------------
// 4) f32 -> bf16 weight convert + transpose:
//    src [E][KD][ND] f32  ->  dst [E][ND][KD] bf16  (column K-runs contiguous)
//    LDS-tiled 64x64, padded stride 80 bf16
// ---------------------------------------------------------------------------
template <int KD, int ND>
__global__ __launch_bounds__(256) void cvtT_kernel(const float* __restrict__ src,
                                                   __bf16* __restrict__ dst)
{
    const int e     = blockIdx.z;
    const int ktile = blockIdx.y * 64;
    const int ntile = blockIdx.x * 64;
    const int t     = threadIdx.x;

    __shared__ __align__(16) __bf16 lT[64 * 80];   // [k][n], stride 80

    const float* s = src + (size_t)e * KD * ND;
    __bf16*      d = dst + (size_t)e * ND * KD;

    // load 64(k) x 64(n) tile, convert, store rows into LDS
    {
        const int kr = t >> 2;          // 0..63
        const int ns = (t & 3) << 4;    // 0,16,32,48
        const float4* sp = (const float4*)(s + (size_t)(ktile + kr) * ND + ntile + ns);
        union { __bf16 h[16]; uint4 q[2]; } tmp;
#pragma unroll
        for (int i = 0; i < 4; ++i) {
            const float4 v = sp[i];
            tmp.h[i * 4 + 0] = (__bf16)v.x;
            tmp.h[i * 4 + 1] = (__bf16)v.y;
            tmp.h[i * 4 + 2] = (__bf16)v.z;
            tmp.h[i * 4 + 3] = (__bf16)v.w;
        }
        uint4* lp = (uint4*)(&lT[kr * 80 + ns]);
        lp[0] = tmp.q[0];
        lp[1] = tmp.q[1];
    }
    __syncthreads();
    // write transposed: 16 contiguous k per thread for one n column
    {
        const int nc = t >> 2;          // 0..63
        const int ks = (t & 3) << 4;    // 0,16,32,48
        union { __bf16 h[16]; uint4 q[2]; } o;
#pragma unroll
        for (int i = 0; i < 16; ++i)
            o.h[i] = lT[(ks + i) * 80 + nc];
        uint4* op = (uint4*)(d + (size_t)(ntile + nc) * KD + ktile + ks);
        op[0] = o.q[0];
        op[1] = o.q[1];
    }
}

// ---------------------------------------------------------------------------
// 5/6) WMMA grouped GEMM:  C[e] = A[e] (MROWS x KD)  *  B[e] (KD x ND)
//   B supplied transposed [E][ND][KD]; both A and B fragments are straight
//   2x global_load_b128 per lane (contiguous 16B runs) -> no LDS, no barriers.
//   block = 256 threads = 8 waves; block tile = 32(M) x 512(N);
//   wave tile = 32 x 64 -> 2 A frags + 4 B frags, 8 WMMAs per k-step.
//   MODE 0: bias + exact GELU -> bf16; MODE 1: bias -> f32.
// ---------------------------------------------------------------------------
template <int KD, int ND, int MODE>
__global__ __launch_bounds__(256) void wmma_gemm_kernel(
    const __bf16* __restrict__ A,     // [E][MROWS][KD]
    const __bf16* __restrict__ BT,    // [E][ND][KD]
    const float*  __restrict__ bias,  // [E][ND]
    void* __restrict__ outp)
{
    const int e     = blockIdx.z;
    const int mtile = blockIdx.y;          // 32 rows
    const int nblk  = blockIdx.x;          // 512 cols
    const int t     = threadIdx.x;
    const int wave  = t >> 5;
    const int lane  = t & 31;
    const int mrow  = lane & 15;
    const int khalf = lane >> 4;

    const int n0 = nblk * 512 + wave * 64;

    // per-lane row pointers for the two A fragments (rows mtile*32 + mrow (+16))
    const __bf16* arow0 =
        A + ((size_t)e * MROWS + (size_t)mtile * 32 + mrow) * KD + khalf * 8;
    const __bf16* arow1 = arow0 + (size_t)16 * KD;
    // per-lane column pointer for B fragment 0 (column n0 + mrow)
    const __bf16* bcol =
        BT + ((size_t)e * ND + (size_t)n0 + mrow) * KD + khalf * 8;

    v8f acc00 = {0.f,0.f,0.f,0.f,0.f,0.f,0.f,0.f};
    v8f acc01 = acc00, acc02 = acc00, acc03 = acc00;
    v8f acc10 = acc00, acc11 = acc00, acc12 = acc00, acc13 = acc00;

    for (int kk = 0; kk < KD; kk += 32) {
        // A fragments: 16-bit A 16x32 layout == two contiguous 16B runs per lane
        union { v16bf v; uint4 q[2]; } a0, a1;
        {
            const uint4* p0 = (const uint4*)(arow0 + kk);
            a0.q[0] = p0[0];            // k run [kk+khalf*8, +8)
            a0.q[1] = p0[2];            // k run [kk+16+khalf*8, +8)
            const uint4* p1 = (const uint4*)(arow1 + kk);
            a1.q[0] = p1[0];
            a1.q[1] = p1[2];
        }
        // B fragments: column K-runs are contiguous in BT -> 2x b128 each
        union { v16bf v; uint4 q[2]; } b0, b1, b2, b3;
        {
            const uint4* q0 = (const uint4*)(bcol + kk);
            const uint4* q1 = (const uint4*)(bcol + (size_t)16 * KD + kk);
            const uint4* q2 = (const uint4*)(bcol + (size_t)32 * KD + kk);
            const uint4* q3 = (const uint4*)(bcol + (size_t)48 * KD + kk);
            b0.q[0] = q0[0]; b0.q[1] = q0[2];
            b1.q[0] = q1[0]; b1.q[1] = q1[2];
            b2.q[0] = q2[0]; b2.q[1] = q2[2];
            b3.q[0] = q3[0]; b3.q[1] = q3[2];
            if (kk + 32 < KD)
                __builtin_prefetch(bcol + kk + 32, 0, 3);
        }

        acc00 = __builtin_amdgcn_wmma_f32_16x16x32_bf16(
            false, a0.v, false, b0.v, (short)0, acc00, false, false);
        acc01 = __builtin_amdgcn_wmma_f32_16x16x32_bf16(
            false, a0.v, false, b1.v, (short)0, acc01, false, false);
        acc02 = __builtin_amdgcn_wmma_f32_16x16x32_bf16(
            false, a0.v, false, b2.v, (short)0, acc02, false, false);
        acc03 = __builtin_amdgcn_wmma_f32_16x16x32_bf16(
            false, a0.v, false, b3.v, (short)0, acc03, false, false);
        acc10 = __builtin_amdgcn_wmma_f32_16x16x32_bf16(
            false, a1.v, false, b0.v, (short)0, acc10, false, false);
        acc11 = __builtin_amdgcn_wmma_f32_16x16x32_bf16(
            false, a1.v, false, b1.v, (short)0, acc11, false, false);
        acc12 = __builtin_amdgcn_wmma_f32_16x16x32_bf16(
            false, a1.v, false, b2.v, (short)0, acc12, false, false);
        acc13 = __builtin_amdgcn_wmma_f32_16x16x32_bf16(
            false, a1.v, false, b3.v, (short)0, acc13, false, false);
    }

    // epilogue: D VGPR r -> row (khalf*8 + r); fragment f column = n0+f*16+mrow
    v8f accs[2][4] = {{acc00, acc01, acc02, acc03},
                      {acc10, acc11, acc12, acc13}};
#pragma unroll
    for (int i = 0; i < 2; ++i) {
#pragma unroll
        for (int f = 0; f < 4; ++f) {
            const int gcol = n0 + f * 16 + mrow;
            const float bv = bias[(size_t)e * ND + gcol];
#pragma unroll
            for (int r = 0; r < 8; ++r) {
                const int grow = mtile * 32 + i * 16 + (khalf << 3) + r;
                float v = accs[i][f][r] + bv;
                if (MODE == 0) {
                    v = 0.5f * v * (1.0f + erff(v * 0.70710678118654752f));
                    ((__bf16*)outp)[((size_t)e * MROWS + grow) * ND + gcol] =
                        (__bf16)v;
                } else {
                    ((float*)outp)[((size_t)e * MROWS + grow) * ND + gcol] = v;
                }
            }
        }
    }
}

// ---------------------------------------------------------------------------
// 7) combine: per-token gather of its <=2 expert rows weighted by gates
// ---------------------------------------------------------------------------
__global__ __launch_bounds__(256) void combine_kernel(
    const float* __restrict__ eo, const int* __restrict__ slot1,
    const int* __restrict__ slot2, const float* __restrict__ g1,
    const float* __restrict__ g2, float* __restrict__ out)
{
    const int tok = blockIdx.x;
    const int t   = threadIdx.x;   // 256 float4 covers D=1024
    const int s1 = slot1[tok];
    const int s2 = slot2[tok];
    float4 res = {0.f, 0.f, 0.f, 0.f};
    if (s1 >= 0) {
        const float w = g1[tok];
        const float4 v = ((const float4*)(eo + (size_t)s1 * DD))[t];
        res.x += w * v.x; res.y += w * v.y; res.z += w * v.z; res.w += w * v.w;
    }
    if (s2 >= 0) {
        const float w = g2[tok];
        const float4 v = ((const float4*)(eo + (size_t)s2 * DD))[t];
        res.x += w * v.x; res.y += w * v.y; res.z += w * v.z; res.w += w * v.w;
    }
    ((float4*)(out + (size_t)tok * DD))[t] = res;
}

// ---------------------------------------------------------------------------
// 8) aux load-balancing loss
// ---------------------------------------------------------------------------
__global__ void loss_kernel(const float* __restrict__ sumraw,
                            const int* __restrict__ cnt, float* __restrict__ loss_out) {
    if (threadIdx.x == 0 && blockIdx.x == 0) {
        float s = 0.f;
        for (int i = 0; i < BB * EE; ++i) s += sumraw[i] * (float)cnt[i];
        // mean(proxy*density)*E*E*coef == sum * coef / N^2
        loss_out[0] = s * (0.01f / ((float)NN * (float)NN));
    }
}

// ---------------------------------------------------------------------------
extern "C" void kernel_launch(void* const* d_in, const int* in_sizes, int n_in,
                              void* d_out, int out_size, void* d_ws, size_t ws_size,
                              hipStream_t stream)
{
    const float* x  = (const float*)d_in[0];
    const float* wg = (const float*)d_in[1];
    const float* w1 = (const float*)d_in[2];
    const float* b1 = (const float*)d_in[3];
    const float* w2 = (const float*)d_in[4];
    const float* b2 = (const float*)d_in[5];
    const float* rp = (const float*)d_in[6];

    float* out = (float*)d_out;
    float* loss_out = out + (size_t)BB * NN * DD;

    char* p = (char*)d_ws;
    auto carve = [&](size_t bytes) -> void* {
        void* r = (void*)p;
        p += (bytes + 255) & ~(size_t)255;
        return r;
    };

    int*    e1o    = (int*)carve((size_t)BB * NN * 4);
    int*    e2o    = (int*)carve((size_t)BB * NN * 4);
    float*  g1o    = (float*)carve((size_t)BB * NN * 4);
    float*  g2o    = (float*)carve((size_t)BB * NN * 4);
    int*    keep2o = (int*)carve((size_t)BB * NN * 4);
    int*    slot1  = (int*)carve((size_t)BB * NN * 4);
    int*    slot2  = (int*)carve((size_t)BB * NN * 4);
    float*  sumraw = (float*)carve((size_t)BB * EE * 4);
    int*    cnt1t  = (int*)carve((size_t)BB * EE * 4);
    int*    slotT  = (int*)carve((size_t)EE * BB * CAP * 4);
    __bf16* ei     = (__bf16*)carve((size_t)EE * MROWS * DD * 2);
    __bf16* hbuf   = (__bf16*)carve((size_t)EE * MROWS * HH * 2);
    __bf16* w1bT   = (__bf16*)carve((size_t)EE * DD * HH * 2);   // [E][H][D]
    __bf16* w2bT   = (__bf16*)carve((size_t)EE * HH * DD * 2);   // [E][D][H]
    float*  eo     = (float*)carve((size_t)EE * MROWS * DD * 4);

    init_kernel<<<(EE * BB * CAP) / 256, 256, 0, stream>>>(sumraw, slotT);

    gating_kernel<<<(BB * NN) / 8, 256, 0, stream>>>(
        x, wg, rp, e1o, e2o, g1o, g2o, keep2o, slot1, slot2, sumraw);

    scan_kernel<<<1, 64, 0, stream>>>(e1o, e2o, keep2o, slot1, slot2, slotT, cnt1t);

    dispatch_kernel<<<EE * BB * CAP, 256, 0, stream>>>(x, slotT, ei);

    // w1 [E][D][H] -> w1bT [E][H][D];  w2 [E][H][D] -> w2bT [E][D][H]
    cvtT_kernel<DD, HH><<<dim3(HH / 64, DD / 64, EE), 256, 0, stream>>>(w1, w1bT);
    cvtT_kernel<HH, DD><<<dim3(DD / 64, HH / 64, EE), 256, 0, stream>>>(w2, w2bT);

    wmma_gemm_kernel<DD, HH, 0>
        <<<dim3(HH / 512, MROWS / 32, EE), 256, 0, stream>>>(ei, w1bT, b1, (void*)hbuf);

    wmma_gemm_kernel<HH, DD, 1>
        <<<dim3(DD / 512, MROWS / 32, EE), 256, 0, stream>>>(hbuf, w2bT, b2, (void*)eo);

    combine_kernel<<<BB * NN, 256, 0, stream>>>(eo, slot1, slot2, g1o, g2o, out);

    loss_kernel<<<1, 1, 0, stream>>>(sumraw, cnt1t, loss_out);
}
